// TransformerBasedContext_18270790877211
// MI455X (gfx1250) — matrix operand order repaired and checked by reading, not compiled, not run on hardware
//
#include <hip/hip_runtime.h>
#include <hip/hip_bf16.h>
#include <math.h>

typedef __bf16 bf16_t;
typedef __attribute__((ext_vector_type(16))) __bf16 v16bf;
typedef __attribute__((ext_vector_type(8)))  __bf16 v8bf;
typedef __attribute__((ext_vector_type(8)))  float  v8f;

#define BB  32
#define LL  512
#define DD  512
#define HH  8
#define DHD 64
#define FFD 2048
#define MM  (BB * LL)

#define EPI_GATE     0
#define EPI_BIAS_BF  1
#define EPI_BIAS_RES 2
#define EPI_GELU_BF  3
#define EPI_GELU_RES 4

__device__ __forceinline__ v8f wmma_bf16(v16bf a, v16bf b, v8f c) {
  // v_wmma_f32_16x16x32_bf16: (neg_a, A, neg_b, B, c_mod, C, reuse_a, reuse_b)
  return __builtin_amdgcn_wmma_f32_16x16x32_bf16(false, a, false, b, (short)0, c,
                                                 false, false);
}

// A-fragment (16x32 bf16): per lane, 16 contiguous K values at 16*lhalf
__device__ __forceinline__ v16bf load_a16(const bf16_t* p) {
  return *(const v16bf*)(p);
}
// B-fragment (32x16 bf16): per lane, 16 contiguous K values at 16*lhalf
__device__ __forceinline__ v16bf load_b16(const bf16_t* p) {
  return *(const v16bf*)(p);
}

__device__ __forceinline__ float gelu_tanh(float x) {
  const float c = 0.7978845608028654f * (x + 0.044715f * x * x * x);
  return 0.5f * x * (1.0f + tanhf(c));
}

// ---------------------------------------------------------------- f32 -> bf16
// 8 elements per thread: 2x b128 in, 1x b128 out (HBM-bandwidth bound pass).
__global__ void f2bf_kernel(const float* __restrict__ in, bf16_t* __restrict__ out,
                            const int n) {
  const int i = (blockIdx.x * 256 + threadIdx.x) * 8;
  if (i < n) {
    const float4 lo = *(const float4*)(in + i);
    const float4 hi = *(const float4*)(in + i + 4);
    v8bf o;
    o[0] = (bf16_t)lo.x; o[1] = (bf16_t)lo.y; o[2] = (bf16_t)lo.z; o[3] = (bf16_t)lo.w;
    o[4] = (bf16_t)hi.x; o[5] = (bf16_t)hi.y; o[6] = (bf16_t)hi.z; o[7] = (bf16_t)hi.w;
    *(v8bf*)(out + i) = o;
  }
}

// -------------------------------------------------------------- WMMA GEMM
// C[M,N] = A[M,K] (bf16) * W[N,K]^T (bf16), fused epilogues.
// block = 128 threads (4 waves), macro-tile 128x128, wave tile 64x64:
// per 32-wide k-step each wave does 16 WMMAs for 16 b128 loads (reuse 4).
template <int EPI>
__global__ __launch_bounds__(128) void gemm_wmma(
    const bf16_t* __restrict__ A, const bf16_t* __restrict__ W,
    const float* __restrict__ bias, const float* __restrict__ res,
    const float* __restrict__ spk, const float scale,
    float* __restrict__ outf, bf16_t* __restrict__ outb,
    const int N, const int K) {
  const int tid   = threadIdx.x;
  const int w     = tid >> 5;
  const int lane  = tid & 31;
  const int lhalf = lane >> 4;
  const int lcol  = lane & 15;
  const int kb8   = lhalf * 8;
  const int kb16  = lhalf * 16;
  const int m0 = blockIdx.y * 128 + (w >> 1) * 64;
  const int n0 = blockIdx.x * 128 + (w & 1) * 64;

  v8f acc[4][4];
#pragma unroll
  for (int i = 0; i < 4; ++i)
#pragma unroll
    for (int j = 0; j < 4; ++j)
#pragma unroll
      for (int r = 0; r < 8; ++r) acc[i][j][r] = 0.f;

  const bf16_t* aRow = A + (size_t)(m0 + lcol) * K + kb16;
  const bf16_t* bRow = W + (size_t)(n0 + lcol) * K + kb16;
  const size_t rowStep = (size_t)16 * K;

  for (int kk = 0; kk < K; kk += 32) {
    v16bf fa[4], fb[4];
#pragma unroll
    for (int mf = 0; mf < 4; ++mf) fa[mf] = load_a16(aRow + (size_t)mf * rowStep + kk);
#pragma unroll
    for (int nf = 0; nf < 4; ++nf) fb[nf] = load_b16(bRow + (size_t)nf * rowStep + kk);
#pragma unroll
    for (int mf = 0; mf < 4; ++mf)
#pragma unroll
      for (int nf = 0; nf < 4; ++nf)
        acc[mf][nf] = wmma_bf16(fa[mf], fb[nf], acc[mf][nf]);
  }

#pragma unroll
  for (int mf = 0; mf < 4; ++mf)
#pragma unroll
    for (int nf = 0; nf < 4; ++nf)
#pragma unroll
      for (int r = 0; r < 8; ++r) {
        const int row = m0 + mf * 16 + lhalf * 8 + r;
        const int col = n0 + nf * 16 + lcol;
        const size_t idx = (size_t)row * N + col;
        const float vv = acc[mf][nf][r];
        if (EPI == EPI_GATE) {
          // sigmoid(x Wg^T) * x + speaker + sinusoidal PE ; emit f32 + bf16
          const float xv = res[idx];
          const float g = 1.f / (1.f + expf(-vv));
          const int l = row & (LL - 1);
          const float freq = expf((float)(col & ~1) * (-0.017988946039015984f));
          const float ang = (float)l * freq;
          const float pe = (col & 1) ? cosf(ang) : sinf(ang);
          const float o = g * xv + spk[idx] + pe;
          outf[idx] = o;
          outb[idx] = (bf16_t)o;
        } else if (EPI == EPI_BIAS_BF) {
          outb[idx] = (bf16_t)((vv + bias[col]) * scale);
        } else if (EPI == EPI_BIAS_RES) {
          outf[idx] = vv + bias[col] + res[idx];   // h = attn_out + x (in place)
        } else if (EPI == EPI_GELU_BF) {
          outb[idx] = (bf16_t)gelu_tanh(vv + bias[col]);
        } else {  // EPI_GELU_RES
          outf[idx] = gelu_tanh(vv + bias[col]) + res[idx];
        }
      }
}

// ------------------------------------------------------------ flash attention
// grid = (L/64, B*H), block = 128 (4 waves); wave handles 16 queries, DH=64.
__global__ __launch_bounds__(128) void flash_attn_kernel(
    const bf16_t* __restrict__ q, const bf16_t* __restrict__ k,
    const bf16_t* __restrict__ v, const int* __restrict__ mask,
    bf16_t* __restrict__ ctx) {
  __shared__ __align__(32) bf16_t Vt[DHD * 64];     // Vt[dh][key], transposed V tile
  __shared__ __align__(32) bf16_t Pt[4][16 * 64];   // per-wave P[m][key]

  const int tid   = threadIdx.x;
  const int w     = tid >> 5;
  const int lane  = tid & 31;
  const int lhalf = lane >> 4;
  const int lcol  = lane & 15;
  const int kb8   = lhalf * 8;
  const int kb16  = lhalf * 16;
  const int bh = blockIdx.y;
  const int b  = bh >> 3;           // H == 8
  const int h  = bh & (HH - 1);
  const int q0 = blockIdx.x * 64 + w * 16;
  const size_t base = ((size_t)b * LL) * DD + (size_t)h * DHD;

  const bf16_t* qrow = q + base + (size_t)(q0 + lcol) * DD;
  const v16bf qa0 = load_a16(qrow + kb16);
  const v16bf qa1 = load_a16(qrow + 32 + kb16);

  float rowMax[8], rowSum[8];
  v8f ofr[4];
#pragma unroll
  for (int r = 0; r < 8; ++r) { rowMax[r] = -3.0e38f; rowSum[r] = 0.f; }
#pragma unroll
  for (int g = 0; g < 4; ++g)
#pragma unroll
    for (int r = 0; r < 8; ++r) ofr[g][r] = 0.f;

  for (int t = 0; t < LL / 64; ++t) {
    const int key0 = t * 64;
    __syncthreads();  // previous iteration's Vt readers done
    // stage V tile transposed: 64 keys x 64 dh
    for (int c = tid; c < 512; c += 128) {
      const int key = c >> 3, dh0 = (c & 7) * 8;
      const v8bf vv = *(const v8bf*)(v + base + (size_t)(key0 + key) * DD + dh0);
#pragma unroll
      for (int j = 0; j < 8; ++j) Vt[(dh0 + j) * 64 + key] = vv[j];
    }
    __syncthreads();

    // S = Q K^T for 16 queries x 64 keys (4 fragments along keys)
    float sreg[4][8];
#pragma unroll
    for (int nf = 0; nf < 4; ++nf) {
      const int key = key0 + nf * 16 + lcol;
      const bf16_t* krow = k + base + (size_t)key * DD;
      const v16bf fb0 = load_b16(krow + kb16);
      const v16bf fb1 = load_b16(krow + 32 + kb16);
      v8f s;
#pragma unroll
      for (int r = 0; r < 8; ++r) s[r] = 0.f;
      s = wmma_bf16(qa0, fb0, s);
      s = wmma_bf16(qa1, fb1, s);
      const bool dead = (mask[b * LL + key] == 0);
#pragma unroll
      for (int r = 0; r < 8; ++r) sreg[nf][r] = dead ? -1.0e10f : s[r];
    }

    // online softmax: row stats via half-wave shuffles (row lives in 16 lanes)
    float sc[8];
#pragma unroll
    for (int r = 0; r < 8; ++r) {
      float mx = fmaxf(fmaxf(sreg[0][r], sreg[1][r]), fmaxf(sreg[2][r], sreg[3][r]));
#pragma unroll
      for (int m = 1; m < 16; m <<= 1) mx = fmaxf(mx, __shfl_xor(mx, m, 32));
      const float nm = fmaxf(rowMax[r], mx);
      sc[r] = __expf(rowMax[r] - nm);
      float ps = 0.f;
#pragma unroll
      for (int nf = 0; nf < 4; ++nf) {
        const float p = __expf(sreg[nf][r] - nm);
        sreg[nf][r] = p;
        ps += p;
      }
#pragma unroll
      for (int m = 1; m < 16; m <<= 1) ps += __shfl_xor(ps, m, 32);
      rowSum[r] = rowSum[r] * sc[r] + ps;
      rowMax[r] = nm;
    }
#pragma unroll
    for (int g = 0; g < 4; ++g)
#pragma unroll
      for (int r = 0; r < 8; ++r) ofr[g][r] *= sc[r];

    // P: C-fragment layout -> LDS -> A-fragment layout (wave-private, DS in-order)
    bf16_t* Pw = &Pt[w][0];
#pragma unroll
    for (int nf = 0; nf < 4; ++nf)
#pragma unroll
      for (int r = 0; r < 8; ++r)
        Pw[(lhalf * 8 + r) * 64 + nf * 16 + lcol] = (bf16_t)sreg[nf][r];

    const v16bf pa0 = load_a16(Pw + lcol * 64 + kb16);
    const v16bf pa1 = load_a16(Pw + lcol * 64 + 32 + kb16);
#pragma unroll
    for (int g = 0; g < 4; ++g) {
      const v16bf vb0 = load_b16(&Vt[(g * 16 + lcol) * 64 + kb16]);
      const v16bf vb1 = load_b16(&Vt[(g * 16 + lcol) * 64 + 32 + kb16]);
      ofr[g] = wmma_bf16(pa0, vb0, ofr[g]);
      ofr[g] = wmma_bf16(pa1, vb1, ofr[g]);
    }
  }

  // normalize and write ctx as (B, L, H*DH) bf16
#pragma unroll
  for (int g = 0; g < 4; ++g)
#pragma unroll
    for (int r = 0; r < 8; ++r) {
      const int qq = q0 + lhalf * 8 + r;
      const int dh = g * 16 + lcol;
      ctx[base + (size_t)qq * DD + dh] = (bf16_t)(ofr[g][r] / rowSum[r]);
    }
}

// ---------------------------------------------------------------- layer norm
__global__ __launch_bounds__(256) void layernorm_bf16(
    const float* __restrict__ hbuf, const float* __restrict__ gamma,
    const float* __restrict__ beta, bf16_t* __restrict__ outb) {
  const int row = blockIdx.x * 8 + (threadIdx.x >> 5);
  const int lane = threadIdx.x & 31;
  const float* xr = hbuf + (size_t)row * DD;
  float s = 0.f, s2 = 0.f;
  for (int i = lane; i < DD; i += 32) { const float vv = xr[i]; s += vv; s2 += vv * vv; }
#pragma unroll
  for (int m = 1; m < 32; m <<= 1) { s += __shfl_xor(s, m, 32); s2 += __shfl_xor(s2, m, 32); }
  const float mu = s * (1.f / DD);
  const float var = s2 * (1.f / DD) - mu * mu;
  const float rs = rsqrtf(var + 1e-5f);
  for (int i = lane; i < DD; i += 32)
    outb[(size_t)row * DD + i] = (bf16_t)((xr[i] - mu) * rs * gamma[i] + beta[i]);
}

// ------------------------------------------------------------------- driver
extern "C" void kernel_launch(void* const* d_in, const int* in_sizes, int n_in,
                              void* d_out, int out_size, void* d_ws, size_t ws_size,
                              hipStream_t stream) {
  (void)in_sizes; (void)n_in; (void)out_size; (void)ws_size;
  const float* x    = (const float*)d_in[0];
  const int*   mask = (const int*)d_in[1];
  const float* spk  = (const float*)d_in[2];
  const float* Wg   = (const float*)d_in[3];
  const float* Wq   = (const float*)d_in[4];
  const float* bq   = (const float*)d_in[5];
  const float* Wk   = (const float*)d_in[6];
  const float* bk   = (const float*)d_in[7];
  const float* Wv   = (const float*)d_in[8];
  const float* bv   = (const float*)d_in[9];
  const float* Wo   = (const float*)d_in[10];
  const float* bo   = (const float*)d_in[11];
  const float* ln_g = (const float*)d_in[12];
  const float* ln_b = (const float*)d_in[13];
  const float* W1   = (const float*)d_in[14];
  const float* b1   = (const float*)d_in[15];
  const float* W2   = (const float*)d_in[16];
  const float* b2   = (const float*)d_in[17];
  float* out = (float*)d_out;

  char* ws = (char*)d_ws;
  const size_t szMD_bf = (size_t)MM * DD * sizeof(bf16_t);
  bf16_t* xb   = (bf16_t*)(ws + 0 * szMD_bf);
  bf16_t* x1b  = (bf16_t*)(ws + 1 * szMD_bf);
  bf16_t* qb   = (bf16_t*)(ws + 2 * szMD_bf);
  bf16_t* kbuf = (bf16_t*)(ws + 3 * szMD_bf);
  bf16_t* vbuf = (bf16_t*)(ws + 4 * szMD_bf);
  bf16_t* midb = (bf16_t*)(ws + 0);  // aliases xb..kbuf (64 MiB), dead by then
  size_t off = 5 * szMD_bf;
  bf16_t* Wgb = (bf16_t*)(ws + off); off += (size_t)DD * DD * 2;
  bf16_t* Wqb = (bf16_t*)(ws + off); off += (size_t)DD * DD * 2;
  bf16_t* Wkb = (bf16_t*)(ws + off); off += (size_t)DD * DD * 2;
  bf16_t* Wvb = (bf16_t*)(ws + off); off += (size_t)DD * DD * 2;
  bf16_t* Wob = (bf16_t*)(ws + off); off += (size_t)DD * DD * 2;
  bf16_t* W1b = (bf16_t*)(ws + off); off += (size_t)FFD * DD * 2;
  bf16_t* W2b = (bf16_t*)(ws + off); off += (size_t)DD * FFD * 2;
  float*  x1  = (float*)(ws + off);  off += (size_t)MM * DD * 4;  // also h (in-place)
  bf16_t* ctxb = (bf16_t*)(ws + off);                              // also LN output
  bf16_t* lnbf = ctxb;

  const dim3 blk(128);

  // f32 -> bf16 conversions (activations + all weights), 8 elems/thread
  f2bf_kernel<<<(MM * DD / 8 + 255) / 256, 256, 0, stream>>>(x, xb, MM * DD);
  f2bf_kernel<<<(DD * DD / 8 + 255) / 256, 256, 0, stream>>>(Wg, Wgb, DD * DD);
  f2bf_kernel<<<(DD * DD / 8 + 255) / 256, 256, 0, stream>>>(Wq, Wqb, DD * DD);
  f2bf_kernel<<<(DD * DD / 8 + 255) / 256, 256, 0, stream>>>(Wk, Wkb, DD * DD);
  f2bf_kernel<<<(DD * DD / 8 + 255) / 256, 256, 0, stream>>>(Wv, Wvb, DD * DD);
  f2bf_kernel<<<(DD * DD / 8 + 255) / 256, 256, 0, stream>>>(Wo, Wob, DD * DD);
  f2bf_kernel<<<(FFD * DD / 8 + 255) / 256, 256, 0, stream>>>(W1, W1b, FFD * DD);
  f2bf_kernel<<<(DD * FFD / 8 + 255) / 256, 256, 0, stream>>>(W2, W2b, DD * FFD);

  // gate + speaker + positional encoding -> x1 (f32) and x1b (bf16)
  gemm_wmma<EPI_GATE><<<dim3(DD / 128, MM / 128), blk, 0, stream>>>(
      xb, Wgb, nullptr, x, spk, 1.f, x1, x1b, DD, DD);
  // Q (scaled by 1/sqrt(DH)), K, V
  gemm_wmma<EPI_BIAS_BF><<<dim3(DD / 128, MM / 128), blk, 0, stream>>>(
      x1b, Wqb, bq, nullptr, nullptr, 0.125f, nullptr, qb, DD, DD);
  gemm_wmma<EPI_BIAS_BF><<<dim3(DD / 128, MM / 128), blk, 0, stream>>>(
      x1b, Wkb, bk, nullptr, nullptr, 1.f, nullptr, kbuf, DD, DD);
  gemm_wmma<EPI_BIAS_BF><<<dim3(DD / 128, MM / 128), blk, 0, stream>>>(
      x1b, Wvb, bv, nullptr, nullptr, 1.f, nullptr, vbuf, DD, DD);
  // flash attention -> ctx (bf16)
  flash_attn_kernel<<<dim3(LL / 64, BB * HH), blk, 0, stream>>>(
      qb, kbuf, vbuf, mask, ctxb);
  // h = ctx Wo^T + bo + x1  (in place over x1)
  gemm_wmma<EPI_BIAS_RES><<<dim3(DD / 128, MM / 128), blk, 0, stream>>>(
      ctxb, Wob, bo, x1, nullptr, 1.f, x1, nullptr, DD, DD);
  // layernorm(h) -> bf16
  layernorm_bf16<<<MM / 8, 256, 0, stream>>>(x1, ln_g, ln_b, lnbf);
  // mid = gelu(ln W1^T + b1) -> bf16
  gemm_wmma<EPI_GELU_BF><<<dim3(FFD / 128, MM / 128), blk, 0, stream>>>(
      lnbf, W1b, b1, nullptr, nullptr, 1.f, nullptr, midb, FFD, DD);
  // out = gelu(mid W2^T + b2) + h -> f32
  gemm_wmma<EPI_GELU_RES><<<dim3(DD / 128, MM / 128), blk, 0, stream>>>(
      midb, W2b, b2, x1, nullptr, 1.f, out, nullptr, DD, FFD);
}
